// MultiHeadAttention_85701777424532
// MI455X (gfx1250) — matrix operand order repaired and checked by reading, not compiled, not run on hardware
//
#include <hip/hip_runtime.h>

#define D_ 2048
#define T_ 2048
#define H_ 16
#define HD_ 128

typedef __attribute__((ext_vector_type(16))) __bf16 bf16x16;
typedef __attribute__((ext_vector_type(8)))  __bf16 bf16x8;
typedef __attribute__((ext_vector_type(8)))  float  f32x8;

// ---------------------------------------------------------------------------
// WMMA helper (wave32, v_wmma_f32_16x16x32_bf16)
// ---------------------------------------------------------------------------
__device__ __forceinline__ f32x8 wmma_bf16(bf16x16 a, bf16x16 b, f32x8 c) {
  return __builtin_amdgcn_wmma_f32_16x16x32_bf16(false, a, false, b,
                                                 (short)0, c, false, false);
}

// A-matrix fragment: 16x32 bf16 tile, row-major source with leading dim ld
// (elements). Lane l holds row M=l&15; lanes 0-15 K={0..7,16..23},
// lanes 16-31 K={8..15,24..31}.
__device__ __forceinline__ bf16x16 load_frag_a(const __bf16* base, int ld) {
  int lane = threadIdx.x & 31;
  const __bf16* p = base + (lane & 15) * ld + ((lane >> 4) << 3);
  bf16x16 f;
  *((bf16x8*)&f)       = *((const bf16x8*)p);
  *(((bf16x8*)&f) + 1) = *((const bf16x8*)(p + 16));
  return f;
}

// B-matrix fragment: 32x16 bf16 tile from B^T row-major [N][K].
// Lanes 0-15 hold col N=l with K={0..7,16..23}; lanes 16-31 K={8..15,24..31}.
__device__ __forceinline__ bf16x16 load_frag_b(const __bf16* base, int ld) {
  int lane = threadIdx.x & 31;
  const __bf16* p = base + (lane & 15) * ld + ((lane >> 4) << 3);
  bf16x16 f;
  *((bf16x8*)&f)       = *((const bf16x8*)p);
  *(((bf16x8*)&f) + 1) = *((const bf16x8*)(p + 16));
  return f;
}

// ---------------------------------------------------------------------------
// Async global->LDS copy (CDNA5: GLOBAL_LOAD_ASYNC_TO_LDS_B128, ASYNCcnt)
// ---------------------------------------------------------------------------
__device__ __forceinline__ void async_copy_b128_off(const __bf16* gsrc,
                                                    unsigned lds_byte_off) {
  asm volatile("global_load_async_to_lds_b128 %0, %1, off"
               :: "v"(lds_byte_off), "v"(gsrc)
               : "memory");
}

__device__ __forceinline__ void wait_async0() {
  asm volatile("s_wait_asynccnt 0x0" ::: "memory");
}

// ---------------------------------------------------------------------------
// Elementwise converters
// ---------------------------------------------------------------------------
__global__ void cvt_f32_bf16_kernel(const float* __restrict__ in,
                                    __bf16* __restrict__ out, int n) {
  int i = blockIdx.x * blockDim.x + threadIdx.x;
  if (i < n) out[i] = (__bf16)in[i];
}

// in[rows][cols] f32 -> out[cols][rows] bf16
__global__ void cvt_transpose_kernel(const float* __restrict__ in,
                                     __bf16* __restrict__ out,
                                     int rows, int cols) {
  int i = blockIdx.x * blockDim.x + threadIdx.x;
  if (i >= rows * cols) return;
  int r = i / cols, c = i - r * cols;
  out[c * rows + r] = (__bf16)in[i];
}

// ---------------------------------------------------------------------------
// GEMM: C[M][N] f32 = A[M][K] bf16 (row-major) x Bt[N][K] bf16 (row-major).
// Block (8 waves) computes a 128x128 C tile. Per 32-deep K-step, 128x32
// panels of A and Bt are staged into double-buffered LDS with async copies;
// each wave owns a 64x32 sub-tile (8 accumulators, 8 WMMA / K-step).
// Staging addresses are loop-carried: one v_add per pointer per step.
// ---------------------------------------------------------------------------
#define LDA_ 40  // padded row stride (elements) for 32-wide panels, 16B aligned

__global__ __launch_bounds__(256)
void gemm_bf16_nt_kernel(const __bf16* __restrict__ A,
                         const __bf16* __restrict__ Bt,
                         float* __restrict__ C, int M, int N, int K) {
  __shared__ __bf16 lA[2][128 * LDA_];
  __shared__ __bf16 lB[2][128 * LDA_];

  int bn_tiles = N >> 7;
  int bm = blockIdx.x / bn_tiles;
  int bn = blockIdx.x - bm * bn_tiles;
  int tm0 = bm << 7, tn0 = bn << 7;

  int tid  = threadIdx.x;
  int lane = tid & 31;
  int wid  = tid >> 5;
  int wr = wid >> 2;        // 0..1 : 64-row slab
  int wc = wid & 3;         // 0..3 : 32-col slab

  // Staging geometry: thread owns 16B segment sseg of rows srow and srow+64.
  int srow = tid >> 2;
  int sseg = (tid & 3) << 3;
  const __bf16* gA0 = A  + (size_t)(tm0 + srow) * K + sseg;
  const __bf16* gA1 = gA0 + (size_t)64 * K;
  const __bf16* gB0 = Bt + (size_t)(tn0 + srow) * K + sseg;
  const __bf16* gB1 = gB0 + (size_t)64 * K;
  unsigned lAo = (unsigned)(unsigned long long)&lA[0][srow * LDA_ + sseg];
  unsigned lBo = (unsigned)(unsigned long long)&lB[0][srow * LDA_ + sseg];
  const unsigned HIo  = 64u * LDA_ * 2;    // +64 rows (bytes)
  const unsigned BUFo = 128u * LDA_ * 2;   // buffer-1 offset (bytes)

  auto stage = [&](unsigned boff) {
    async_copy_b128_off(gA0, lAo + boff);
    async_copy_b128_off(gA1, lAo + boff + HIo);
    async_copy_b128_off(gB0, lBo + boff);
    async_copy_b128_off(gB1, lBo + boff + HIo);
    gA0 += 32; gA1 += 32; gB0 += 32; gB1 += 32;   // advance one K-step
  };

  f32x8 acc[4][2];
#pragma unroll
  for (int rt = 0; rt < 4; ++rt)
#pragma unroll
    for (int ct = 0; ct < 2; ++ct) acc[rt][ct] = (f32x8){};

  stage(0);
  wait_async0();
  __syncthreads();

  int buf = 0;
  for (int k = 0; k < K; k += 32) {
    if (k + 32 < K) stage((buf ^ 1) ? BUFo : 0u);

    bf16x16 bfr[2];
#pragma unroll
    for (int ct = 0; ct < 2; ++ct)
      bfr[ct] = load_frag_b(&lB[buf][(wc * 32 + ct * 16) * LDA_], LDA_);
#pragma unroll
    for (int rt = 0; rt < 4; ++rt) {
      bf16x16 af = load_frag_a(&lA[buf][(wr * 64 + rt * 16) * LDA_], LDA_);
#pragma unroll
      for (int ct = 0; ct < 2; ++ct)
        acc[rt][ct] = wmma_bf16(af, bfr[ct], acc[rt][ct]);
    }

    wait_async0();
    __syncthreads();
    buf ^= 1;
  }

  int ro  = (lane >> 4) << 3;
  int col = lane & 15;
#pragma unroll
  for (int rt = 0; rt < 4; ++rt) {
#pragma unroll
    for (int ct = 0; ct < 2; ++ct) {
#pragma unroll
      for (int r = 0; r < 8; ++r) {
        int row = tm0 + wr * 64 + rt * 16 + ro + r;
        int cc  = tn0 + wc * 32 + ct * 16 + col;
        C[(size_t)row * N + cc] = acc[rt][ct][r];
      }
    }
  }
}

// ---------------------------------------------------------------------------
// RoPE + convert: Qf/Kf f32 [T][D] -> Qb/Kb bf16 [T][D] (rotated; Q scaled by
// 1/sqrt(HD)); Vf f32 [T][D] -> Vt bf16 [D][T] (transposed for PV B-frags).
// ---------------------------------------------------------------------------
__global__ void rope_cvt_kernel(const float* __restrict__ Qf,
                                const float* __restrict__ Kf,
                                const float* __restrict__ Vf,
                                const float* __restrict__ sin_t,
                                const float* __restrict__ cos_t,
                                __bf16* __restrict__ Qb,
                                __bf16* __restrict__ Kb,
                                __bf16* __restrict__ Vt) {
  int i = blockIdx.x * blockDim.x + threadIdx.x;
  if (i >= T_ * H_ * (HD_ / 2)) return;
  int p = i & 63;
  int h = (i >> 6) & (H_ - 1);
  int t = i >> 10;
  int base = t * D_ + h * HD_;

  float s1 = sin_t[t * HD_ + p],      c1 = cos_t[t * HD_ + p];
  float s2 = sin_t[t * HD_ + p + 64], c2 = cos_t[t * HD_ + p + 64];

  const float qs = 0.08838834764831845f;  // 1/sqrt(128)
  float q1 = Qf[base + p], q2 = Qf[base + p + 64];
  Qb[base + p]      = (__bf16)((q1 * c1 - q2 * s1) * qs);
  Qb[base + p + 64] = (__bf16)((q2 * c2 + q1 * s2) * qs);

  float k1 = Kf[base + p], k2 = Kf[base + p + 64];
  Kb[base + p]      = (__bf16)(k1 * c1 - k2 * s1);
  Kb[base + p + 64] = (__bf16)(k2 * c2 + k1 * s2);

  Vt[(h * HD_ + p) * T_ + t]      = (__bf16)Vf[base + p];
  Vt[(h * HD_ + p + 64) * T_ + t] = (__bf16)Vf[base + p + 64];
}

// ---------------------------------------------------------------------------
// Flash attention: one wave per (head, 16-query tile). Causal + same-doc mask,
// online softmax. Q*K^T and P*V via bf16 WMMA. P re-layout C->A via LDS.
// ---------------------------------------------------------------------------
__global__ __launch_bounds__(256)
void attn_kernel(const __bf16* __restrict__ Qb, const __bf16* __restrict__ Kb,
                 const __bf16* __restrict__ Vt, const int* __restrict__ doc,
                 __bf16* __restrict__ Ab) {
  __shared__ float sP[8][16][33];

  int wid  = threadIdx.x >> 5;
  int lane = threadIdx.x & 31;
  int gw   = blockIdx.x * 8 + wid;
  int qt   = gw & 127;
  int head = gw >> 7;
  int q0   = qt << 4;

  const __bf16* Qh = Qb + head * HD_;
  const __bf16* Kh = Kb + head * HD_;
  const __bf16* Vh = Vt + (size_t)head * HD_ * T_;

  bf16x16 qf[4];
#pragma unroll
  for (int kk = 0; kk < 4; ++kk)
    qf[kk] = load_frag_a(Qh + q0 * D_ + kk * 32, D_);

  float m[8], lsum[8];
  f32x8 acc[8];
#pragma unroll
  for (int r = 0; r < 8; ++r) { m[r] = -1e30f; lsum[r] = 0.f; }
#pragma unroll
  for (int nt = 0; nt < 8; ++nt) acc[nt] = (f32x8){};

  int ro   = (lane >> 4) << 3;
  int ncol = lane & 15;
  int dq[8];
#pragma unroll
  for (int r = 0; r < 8; ++r) dq[r] = doc[q0 + ro + r];

  int nsteps = (q0 + 16 + 31) >> 5;
  for (int js = 0; js < nsteps; ++js) {
    int j0 = js << 5;

    f32x8 s0 = {}, s1 = {};
#pragma unroll
    for (int kk = 0; kk < 4; ++kk) {
      bf16x16 b0 = load_frag_b(Kh + j0 * D_ + kk * 32, D_);
      bf16x16 b1 = load_frag_b(Kh + (j0 + 16) * D_ + kk * 32, D_);
      s0 = wmma_bf16(qf[kk], b0, s0);
      s1 = wmma_bf16(qf[kk], b1, s1);
    }

    int k0g = j0 + ncol, k1g = j0 + 16 + ncol;
    int dk0 = doc[k0g], dk1 = doc[k1g];

#pragma unroll
    for (int r = 0; r < 8; ++r) {
      int qi = q0 + ro + r;
      bool v0 = (k0g <= qi) && (dk0 == dq[r]);
      bool v1 = (k1g <= qi) && (dk1 == dq[r]);
      float x0 = v0 ? s0[r] : -1e30f;
      float x1 = v1 ? s1[r] : -1e30f;
      float mx = fmaxf(x0, x1);
      for (int d = 1; d < 16; d <<= 1) mx = fmaxf(mx, __shfl_xor(mx, d, 32));
      float mn = fmaxf(m[r], mx);
      float p0 = v0 ? __expf(x0 - mn) : 0.f;
      float p1 = v1 ? __expf(x1 - mn) : 0.f;
      float rs = p0 + p1;
      for (int d = 1; d < 16; d <<= 1) rs += __shfl_xor(rs, d, 32);
      float alpha = __expf(m[r] - mn);
      m[r] = mn;
      lsum[r] = lsum[r] * alpha + rs;
#pragma unroll
      for (int nt = 0; nt < 8; ++nt) acc[nt][r] *= alpha;
      sP[wid][ro + r][ncol]      = p0;
      sP[wid][ro + r][16 + ncol] = p1;
    }

    bf16x16 pf;
    {
      int row = lane & 15;
      int kb  = (lane >> 4) << 3;
#pragma unroll
      for (int i = 0; i < 16; ++i) {
        int kidx = ((i >> 3) << 4) + kb + (i & 7);
        pf[i] = (__bf16)sP[wid][row][kidx];
      }
    }

#pragma unroll
    for (int nt = 0; nt < 8; ++nt) {
      bf16x16 vf = load_frag_b(Vh + (nt * 16) * T_ + j0, T_);
      acc[nt] = wmma_bf16(pf, vf, acc[nt]);
    }
  }

#pragma unroll
  for (int nt = 0; nt < 8; ++nt) {
#pragma unroll
    for (int r = 0; r < 8; ++r) {
      float val = acc[nt][r] / lsum[r];
      Ab[(q0 + ro + r) * D_ + head * HD_ + nt * 16 + ncol] = (__bf16)val;
    }
  }
}

// ---------------------------------------------------------------------------
// Launch
// ---------------------------------------------------------------------------
extern "C" void kernel_launch(void* const* d_in, const int* in_sizes, int n_in,
                              void* d_out, int out_size, void* d_ws, size_t ws_size,
                              hipStream_t stream) {
  const float* x    = (const float*)d_in[0];
  const float* Wq   = (const float*)d_in[1];
  const float* Wk   = (const float*)d_in[2];
  const float* Wv   = (const float*)d_in[3];
  const float* Wo   = (const float*)d_in[4];
  const float* sint = (const float*)d_in[5];
  const float* cost = (const float*)d_in[6];
  const int*   doc  = (const int*)d_in[7];
  float* out = (float*)d_out;

  const size_t SZ_BF = (size_t)T_ * D_ * 2;  // 8 MB
  const size_t SZ_F  = (size_t)T_ * D_ * 4;  // 16 MB
  char* w = (char*)d_ws;
  __bf16* Xb  = (__bf16*)(w);
  __bf16* WqT = (__bf16*)(w + 1 * SZ_BF);
  __bf16* WkT = (__bf16*)(w + 2 * SZ_BF);
  __bf16* WvT = (__bf16*)(w + 3 * SZ_BF);
  __bf16* WoT = (__bf16*)(w + 4 * SZ_BF);
  float*  Qf  = (float*)(w + 5 * SZ_BF);
  float*  Kf  = (float*)(w + 5 * SZ_BF + 1 * SZ_F);
  float*  Vf  = (float*)(w + 5 * SZ_BF + 2 * SZ_F);
  __bf16* Qb  = (__bf16*)(w + 5 * SZ_BF + 3 * SZ_F);
  __bf16* Kb  = (__bf16*)(w + 6 * SZ_BF + 3 * SZ_F);
  __bf16* Vt  = (__bf16*)(w + 7 * SZ_BF + 3 * SZ_F);
  __bf16* Ab  = (__bf16*)(w + 8 * SZ_BF + 3 * SZ_F);

  const int n = T_ * D_;
  const int eb = (n + 255) / 256;

  cvt_f32_bf16_kernel<<<eb, 256, 0, stream>>>(x, Xb, n);
  cvt_transpose_kernel<<<eb, 256, 0, stream>>>(Wq, WqT, D_, D_);
  cvt_transpose_kernel<<<eb, 256, 0, stream>>>(Wk, WkT, D_, D_);
  cvt_transpose_kernel<<<eb, 256, 0, stream>>>(Wv, WvT, D_, D_);
  cvt_transpose_kernel<<<eb, 256, 0, stream>>>(Wo, WoT, D_, D_);

  const int gemm_blocks = (T_ / 128) * (D_ / 128);  // 256
  gemm_bf16_nt_kernel<<<gemm_blocks, 256, 0, stream>>>(Xb, WqT, Qf, T_, D_, D_);
  gemm_bf16_nt_kernel<<<gemm_blocks, 256, 0, stream>>>(Xb, WkT, Kf, T_, D_, D_);
  gemm_bf16_nt_kernel<<<gemm_blocks, 256, 0, stream>>>(Xb, WvT, Vf, T_, D_, D_);

  const int rn = T_ * H_ * (HD_ / 2);
  rope_cvt_kernel<<<(rn + 255) / 256, 256, 0, stream>>>(Qf, Kf, Vf, sint, cost,
                                                        Qb, Kb, Vt);

  attn_kernel<<<(H_ * (T_ / 16)) / 8, 256, 0, stream>>>(Qb, Kb, Vt, doc, Ab);

  gemm_bf16_nt_kernel<<<gemm_blocks, 256, 0, stream>>>(Ab, WoT, out, T_, D_, D_);
}
